// BatchedTransformerLayer_68332929679619
// MI455X (gfx1250) — matrix operand order repaired and checked by reading, not compiled, not run on hardware
//
#include <hip/hip_runtime.h>
#include <hip/hip_bf16.h>

typedef __attribute__((ext_vector_type(16))) _Float16 v16h;
typedef __attribute__((ext_vector_type(8)))  _Float16 v8h;
typedef __attribute__((ext_vector_type(8)))  float    v8f;

#define N_NODES 50000
#define N_EDGES 800000
#define D_MODEL 128

// ---------------------------------------------------------------------------
// f32 GEMM via V_WMMA_F32_16X16X32_F16, f32 accumulate.
// C[M x N] = A[M x K] @ B[K x N] (+bias) (+relu) (+resid)
// One wave -> 16x32 C tile (two 16x16 WMMA tiles sharing the A fragment).
// Block = 256 threads = 8 waves (8 M-tiles). grid.x = ceil(M/128), grid.y = N/32.
// B tile (K x 32) is staged once per block into LDS as f16, col-major, so each
// lane's WMMA fragment is two 16-byte runs 32 bytes apart (2x ds_load_b128).
// ---------------------------------------------------------------------------
template <int K, bool BIAS, bool RELU, bool RESID>
__global__ void gemm_wmma(const float* __restrict__ A,
                          const float* __restrict__ B,
                          const float* __restrict__ bias,
                          const float* __restrict__ resid,
                          float* __restrict__ C,
                          int M, int N)
{
    __shared__ _Float16 Bs[K * 32];      // [kstep][col][krel], 32x32 f16 tiles

    const int wave = threadIdx.x >> 5;
    const int lane = threadIdx.x & 31;
    const int half = lane >> 4;          // 0: lanes 0-15, 1: lanes 16-31
    const int r    = lane & 15;
    const int tileM = (blockIdx.x * 8 + wave) * 16;
    const int tileN = blockIdx.y * 32;

    // ---- stage B tile (coalesced: 32 consecutive cols per 32-thread group) ----
#pragma unroll
    for (int idx = threadIdx.x; idx < K * 32; idx += 256) {
        const int krow = idx >> 5;       // 0..K-1
        const int col  = idx & 31;
        const float v  = B[(size_t)krow * N + tileN + col];
        const int kstep = krow >> 5, krel = krow & 31;
        Bs[kstep * 1024 + col * 32 + krel] = (_Float16)v;
    }
    __syncthreads();
    if (tileM >= M) return;              // wave-uniform: EXEC stays all-ones

    v8f acc0 = {}, acc1 = {};
#pragma unroll
    for (int kstep = 0; kstep < K / 32; ++kstep) {
        const int k0 = kstep * 32;
        // A fragment (16x32): lane holds row (tileM + r).
        // elements 0..7  -> K = k0 + half*8 + i
        // elements 8..15 -> K = k0 + 16 + half*8 + i
        const float* Arow = A + (size_t)(tileM + r) * K + k0 + half * 8;
        const float4 f0 = *(const float4*)(Arow + 0);
        const float4 f1 = *(const float4*)(Arow + 4);
        const float4 f2 = *(const float4*)(Arow + 16);
        const float4 f3 = *(const float4*)(Arow + 20);
        const float av[16] = { f0.x, f0.y, f0.z, f0.w,  f1.x, f1.y, f1.z, f1.w,
                               f2.x, f2.y, f2.z, f2.w,  f3.x, f3.y, f3.z, f3.w };
        v16h a;
#pragma unroll
        for (int i = 0; i < 16; ++i) a[i] = (_Float16)av[i];

        // B fragments from LDS: element i<8 -> B[k0 + half*8 + i], i>=8 -> B[k0 + 16 + half*8 + (i-8)]; col tileN + r (+16)
        const _Float16* bp0 = &Bs[kstep * 1024 + r * 32 + half * 8];
        const _Float16* bp1 = &Bs[kstep * 1024 + (r + 16) * 32 + half * 8];
        const v16h b0 = __builtin_shufflevector(
            *(const v8h*)bp0, *(const v8h*)(bp0 + 16),
            0, 1, 2, 3, 4, 5, 6, 7, 8, 9, 10, 11, 12, 13, 14, 15);
        const v16h b1 = __builtin_shufflevector(
            *(const v8h*)bp1, *(const v8h*)(bp1 + 16),
            0, 1, 2, 3, 4, 5, 6, 7, 8, 9, 10, 11, 12, 13, 14, 15);

        acc0 = __builtin_amdgcn_wmma_f32_16x16x32_f16(
            false, a, false, b0, (short)0, acc0, false, false);
        acc1 = __builtin_amdgcn_wmma_f32_16x16x32_f16(
            false, a, false, b1, (short)0, acc1, false, false);
    }

    float bia0 = 0.0f, bia1 = 0.0f;
    if (BIAS) { bia0 = bias[tileN + r]; bia1 = bias[tileN + 16 + r]; }

    // C/D layout: VGPR v -> row v + 8*half, col r
#pragma unroll
    for (int v = 0; v < 8; ++v) {
        const int row = tileM + v + half * 8;
        const size_t base = (size_t)row * N + tileN + r;
        float o0 = acc0[v], o1 = acc1[v];
        if (BIAS)  { o0 += bia0;            o1 += bia1; }
        if (RELU)  { o0 = fmaxf(o0, 0.0f);  o1 = fmaxf(o1, 0.0f); }
        if (RESID) { o0 += resid[base];     o1 += resid[base + 16]; }
        C[base] = o0;
        C[base + 16] = o1;
    }
}

// ---------------------------------------------------------------------------
// Edge attention: one thread per (edge, head).
// score = sum_d K[src]*Q[dst]*E / 4 ; s = exp(clamp(score,-5,5))
// wV[dst] += s * V[src] ; z[dst,h] += s   (f32 global atomics, L2-resident)
// ---------------------------------------------------------------------------
__global__ void edge_attn(const float* __restrict__ Qh,
                          const float* __restrict__ Kh,
                          const float* __restrict__ Vh,
                          const float* __restrict__ Eh,
                          const int* __restrict__ src,
                          const int* __restrict__ dst,
                          float* __restrict__ wV,
                          float* __restrict__ z)
{
    const int idx = blockIdx.x * 256 + threadIdx.x;   // edge*8 + head
    if (idx >= N_EDGES * 8) return;
    const int e  = idx >> 3;
    const int hh = idx & 7;
    const int s  = src[e];
    const int d  = dst[e];

    const float4* kp = (const float4*)(Kh + (size_t)s * D_MODEL + hh * 16);
    const float4* qp = (const float4*)(Qh + (size_t)d * D_MODEL + hh * 16);
    const float4* ep = (const float4*)(Eh + (size_t)e * D_MODEL + hh * 16);

    float sum = 0.0f;
#pragma unroll
    for (int j = 0; j < 4; ++j) {
        const float4 kv = kp[j], qv = qp[j], ev = ep[j];
        sum += kv.x * qv.x * ev.x + kv.y * qv.y * ev.y +
               kv.z * qv.z * ev.z + kv.w * qv.w * ev.w;
    }
    sum *= 0.25f;                                    // 1/sqrt(16)
    sum = fminf(5.0f, fmaxf(-5.0f, sum));
    const float sc = expf(sum);

    const float4* vp  = (const float4*)(Vh + (size_t)s * D_MODEL + hh * 16);
    float*        wvp = wV + (size_t)d * D_MODEL + hh * 16;
#pragma unroll
    for (int j = 0; j < 4; ++j) {
        const float4 vv = vp[j];
        atomicAdd(wvp + 4 * j + 0, vv.x * sc);
        atomicAdd(wvp + 4 * j + 1, vv.y * sc);
        atomicAdd(wvp + 4 * j + 2, vv.z * sc);
        atomicAdd(wvp + 4 * j + 3, vv.w * sc);
    }
    atomicAdd(z + (size_t)d * 8 + hh, sc);
}

// h_attn = wV / (z + 1e-6), in place over wV
__global__ void finalize_attn(float* __restrict__ wV,
                              const float* __restrict__ z, int n)
{
    const int i = blockIdx.x * 256 + threadIdx.x;
    if (i >= n) return;
    const int node = i >> 7;
    const int col  = i & 127;
    wV[i] = wV[i] / (z[(size_t)node * 8 + (col >> 4)] + 1e-6f);
}

// ---------------------------------------------------------------------------
// BatchNorm: per-column atomic partial sums -> finalize -> apply
// ---------------------------------------------------------------------------
__global__ void bn_stats(const float* __restrict__ x,
                         float* __restrict__ sums,
                         float* __restrict__ sumsq, int M)
{
    const int col = threadIdx.x & 127;
    float s = 0.0f, s2 = 0.0f;
    for (int row = blockIdx.x * 2 + (threadIdx.x >> 7); row < M;
         row += gridDim.x * 2) {
        const float v = x[(size_t)row * 128 + col];
        s += v; s2 += v * v;
    }
    __shared__ float ls[256], ls2[256];
    ls[threadIdx.x] = s; ls2[threadIdx.x] = s2;
    __syncthreads();
    if (threadIdx.x < 128) {
        atomicAdd(&sums[col],  ls[threadIdx.x]  + ls[threadIdx.x + 128]);
        atomicAdd(&sumsq[col], ls2[threadIdx.x] + ls2[threadIdx.x + 128]);
    }
}

__global__ void bn_finalize(const float* __restrict__ sums,
                            const float* __restrict__ sumsq,
                            float* __restrict__ mean,
                            float* __restrict__ rstd, float invM)
{
    const int c = threadIdx.x;   // 128 threads
    const float m = sums[c] * invM;
    const float v = sumsq[c] * invM - m * m;
    mean[c] = m;
    rstd[c] = rsqrtf(v + 1e-5f);
}

__global__ void bn_apply(const float* __restrict__ x,
                         const float* __restrict__ mean,
                         const float* __restrict__ rstd,
                         const float* __restrict__ g,
                         const float* __restrict__ b,
                         float* __restrict__ y, int n)
{
    const int i = blockIdx.x * 256 + threadIdx.x;
    if (i >= n) return;
    const int col = i & 127;
    y[i] = (x[i] - mean[col]) * rstd[col] * g[col] + b[col];
}

__global__ void zero_f32(float* __restrict__ p, int n)
{
    const int i = blockIdx.x * 256 + threadIdx.x;
    if (i < n) p[i] = 0.0f;
}

// ---------------------------------------------------------------------------
extern "C" void kernel_launch(void* const* d_in, const int* in_sizes, int n_in,
                              void* d_out, int out_size, void* d_ws, size_t ws_size,
                              hipStream_t stream)
{
    const float* h    = (const float*)d_in[0];
    // d_in[1] = p (unused by reference)
    const float* e    = (const float*)d_in[2];
    const float* Wq   = (const float*)d_in[3];
    const float* Wk   = (const float*)d_in[4];
    const float* We   = (const float*)d_in[5];
    const float* Wv   = (const float*)d_in[6];
    const float* Wo   = (const float*)d_in[7];
    const float* bo   = (const float*)d_in[8];
    const float* bn1g = (const float*)d_in[9];
    const float* bn1b = (const float*)d_in[10];
    const float* bn2g = (const float*)d_in[11];
    const float* bn2b = (const float*)d_in[12];
    const float* W1   = (const float*)d_in[13];
    const float* b1   = (const float*)d_in[14];
    const float* W2   = (const float*)d_in[15];
    const float* b2   = (const float*)d_in[16];
    const int*   src  = (const int*)d_in[17];
    const int*   dst  = (const int*)d_in[18];
    float* out = (float*)d_out;

    // ---- workspace layout (floats) ----
    float* ws = (float*)d_ws;
    const size_t ND = (size_t)N_NODES * D_MODEL;       // 6.4e6
    float* Qh    = ws;                                 // ND
    float* Kh    = Qh   + ND;                          // ND
    float* Vh    = Kh   + ND;                          // ND
    float* wV    = Vh   + ND;                          // ND (becomes h_attn)
    float* h2    = wV   + ND;                          // ND (reused as X later)
    float* h2n   = h2   + ND;                          // ND
    float* ffn1  = h2n  + ND;                          // 2*ND
    float* z     = ffn1 + 2 * ND;                      // N_NODES*8
    float* stats = z    + (size_t)N_NODES * 8;         // 512
    float* sums  = stats;        float* sumsq = stats + 128;
    float* mean  = stats + 256;  float* rstd  = stats + 384;
    float* Eh    = stats + 512;                        // E*128

    const dim3 blk(256);
    const dim3 gN((N_NODES / 16 + 7) / 8, D_MODEL / 32);       // M=50000, N=128
    const dim3 gE((N_EDGES / 16 + 7) / 8, D_MODEL / 32);       // M=800000, N=128
    const dim3 gF1((N_NODES / 16 + 7) / 8, 2 * D_MODEL / 32);  // N=256

    // zero accumulators
    zero_f32<<<(int)((ND + 255) / 256), blk, 0, stream>>>(wV, (int)ND);
    zero_f32<<<(N_NODES * 8 + 255) / 256, blk, 0, stream>>>(z, N_NODES * 8);

    // projections (K=128, no bias/relu/resid)
    gemm_wmma<128, false, false, false><<<gN, blk, 0, stream>>>(
        h, Wq, nullptr, nullptr, Qh, N_NODES, D_MODEL);
    gemm_wmma<128, false, false, false><<<gN, blk, 0, stream>>>(
        h, Wk, nullptr, nullptr, Kh, N_NODES, D_MODEL);
    gemm_wmma<128, false, false, false><<<gN, blk, 0, stream>>>(
        h, Wv, nullptr, nullptr, Vh, N_NODES, D_MODEL);
    gemm_wmma<128, false, false, false><<<gE, blk, 0, stream>>>(
        e, We, nullptr, nullptr, Eh, N_EDGES, D_MODEL);

    // per-edge attention + scatter
    edge_attn<<<(N_EDGES * 8 + 255) / 256, blk, 0, stream>>>(Qh, Kh, Vh, Eh,
                                                             src, dst, wV, z);
    finalize_attn<<<(int)((ND + 255) / 256), blk, 0, stream>>>(wV, z, (int)ND);

    // output projection + residual: h2 = h + h_attn @ Wo + bo
    gemm_wmma<128, true, false, true><<<gN, blk, 0, stream>>>(
        wV, Wo, bo, h, h2, N_NODES, D_MODEL);

    // BN1
    zero_f32<<<1, blk, 0, stream>>>(stats, 256);
    bn_stats<<<512, blk, 0, stream>>>(h2, sums, sumsq, N_NODES);
    bn_finalize<<<1, 128, 0, stream>>>(sums, sumsq, mean, rstd, 1.0f / N_NODES);
    bn_apply<<<(int)((ND + 255) / 256), blk, 0, stream>>>(h2, mean, rstd,
                                                          bn1g, bn1b, h2n, (int)ND);

    // FFN: ffn1 = relu(h2n @ W1 + b1)   [N=256]
    gemm_wmma<128, true, true, false><<<gF1, blk, 0, stream>>>(
        h2n, W1, b1, nullptr, ffn1, N_NODES, 2 * D_MODEL);
    // X = h2n + ffn1 @ W2 + b2   (reuse h2 buffer)
    gemm_wmma<256, true, false, true><<<gN, blk, 0, stream>>>(
        ffn1, W2, b2, h2n, h2, N_NODES, D_MODEL);

    // BN2 -> out
    zero_f32<<<1, blk, 0, stream>>>(stats, 256);
    bn_stats<<<512, blk, 0, stream>>>(h2, sums, sumsq, N_NODES);
    bn_finalize<<<1, 128, 0, stream>>>(sums, sumsq, mean, rstd, 1.0f / N_NODES);
    bn_apply<<<(int)((ND + 255) / 256), blk, 0, stream>>>(h2, mean, rstd,
                                                          bn2g, bn2b, out, (int)ND);
}
